// GNNLayer_1984274891278
// MI455X (gfx1250) — matrix operand order repaired and checked by reading, not compiled, not run on hardware
//
#include <hip/hip_runtime.h>
#include <hip/hip_bf16.h>

typedef __attribute__((ext_vector_type(16))) __bf16 v16bf;
typedef __attribute__((ext_vector_type(8)))  __bf16 v8bf;
typedef __attribute__((ext_vector_type(8)))  float  v8f;

#define NODE_DIM 128
#define HID      128
#define HID2     256
#define TILE_E   128

// ---------------------------------------------------------------------------
// Fragment helpers (layouts per CDNA5 ISA 7.12.2, wave32)
// ---------------------------------------------------------------------------
__device__ __forceinline__ v16bf frag_from_chunks(v8bf lo, v8bf hi) {
    v16bf r;
#pragma unroll
    for (int i = 0; i < 8; ++i) { r[i] = lo[i]; r[i + 8] = hi[i]; }
    return r;
}

// A fragment: 16x32 bf16 tile at (m0, k0) of a row-major [rows][rowStride] array.
// lane holds row m0+(lane&15); K chunks [k0+8*half, +8) and [k0+16+8*half, +8).
__device__ __forceinline__ v16bf load_frag_a(const __bf16* base, int rowStride,
                                             int m0, int k0, int lane) {
    const int m    = m0 + (lane & 15);
    const int half = lane >> 4;
    const __bf16* p = base + m * rowStride + k0 + 8 * half;
    v8bf lo = *(const v8bf*)p;
    v8bf hi = *(const v8bf*)(p + 16);
    return frag_from_chunks(lo, hi);
}

// B fragment: 32x16 tile at (k0, n0) of weights stored transposed as wt[n*K + k].
// lane holds column n0+(lane&15); K chunks [k0+8*half, +8) and [k0+16+8*half, +8).
__device__ __forceinline__ v16bf load_frag_b(const __bf16* wt, int K,
                                             int n0, int k0, int lane) {
    const int n    = n0 + (lane & 15);
    const int half = lane >> 4;
    const __bf16* p = wt + n * K + k0 + 8 * half;
    v8bf lo = *(const v8bf*)p;
    v8bf hi = *(const v8bf*)(p + 16);
    return frag_from_chunks(lo, hi);
}

// ---------------------------------------------------------------------------
// Prep: W1/W2/W3 fp32 [K][N]  ->  bf16 transposed [N][K] in workspace
// ---------------------------------------------------------------------------
__global__ void gnn_prep_weights(const float* __restrict__ W1,
                                 const float* __restrict__ W2,
                                 const float* __restrict__ W3,
                                 __bf16* __restrict__ w1t,
                                 __bf16* __restrict__ w2t,
                                 __bf16* __restrict__ w3t) {
    const int i = blockIdx.x * blockDim.x + threadIdx.x;
    if (i < 128 * 128) {               // W1: K=128, N=128
        const int n = i >> 7, k = i & 127;
        w1t[n * 128 + k] = (__bf16)W1[k * 128 + n];
    }
    if (i < 256 * 128) {               // W2: K=128, N=256
        const int n = i >> 7, k = i & 127;
        w2t[n * 128 + k] = (__bf16)W2[k * 256 + n];
    }
    if (i < 128 * 256) {               // W3: K=256, N=128
        const int n = i >> 8, k = i & 255;
        w3t[n * 256 + k] = (__bf16)W3[k * 128 + n];
    }
}

// ---------------------------------------------------------------------------
// Init: out = h  (atomics accumulate deltas on top)
// ---------------------------------------------------------------------------
__global__ void gnn_init_out(const float4* __restrict__ h4,
                             float4* __restrict__ out4, int n4) {
    int i = blockIdx.x * blockDim.x + threadIdx.x;
    const int stride = gridDim.x * blockDim.x;
    for (; i < n4; i += stride) out4[i] = h4[i];
}

// ---------------------------------------------------------------------------
// Main fused kernel: 128 edges / block, 8 wave32, whole MLP in LDS + WMMA
// ---------------------------------------------------------------------------
__global__ void __launch_bounds__(256, 1)
gnn_edge_mlp(const float* __restrict__ h, const int* __restrict__ rowidx,
             const __bf16* __restrict__ w1t, const __bf16* __restrict__ w2t,
             const __bf16* __restrict__ w3t,
             const float* __restrict__ b1, const float* __restrict__ b2,
             const float* __restrict__ b3,
             float* __restrict__ out, int nEdges) {
    extern __shared__ char smembase[];
    __bf16* hA = (__bf16*)smembase;                 // [128][128] gathered bf16
    __bf16* x1 = hA + TILE_E * HID;                 // [128][128]
    __bf16* x2 = x1 + TILE_E * HID;                 // [128][256]
    int* srows = (int*)(x2 + TILE_E * HID2);        // [128]

    const int tid  = threadIdx.x;
    const int lane = tid & 31;
    const int wave = tid >> 5;
    const int e0   = blockIdx.x * TILE_E;

    // ---- stage row indices --------------------------------------------------
    if (tid < TILE_E) {
        const int e = e0 + tid;
        srows[tid] = (e < nEdges) ? rowidx[e] : 0;
    }
    __syncthreads();

    // ---- gather h[row] -> bf16 LDS (2 threads per edge row) ----------------
    {
        const int s  = tid >> 1;
        const int hf = tid & 1;
        const float4* src = (const float4*)(h + (size_t)srows[s] * NODE_DIM + 64 * hf);
        __bf16* dst = hA + s * NODE_DIM + 64 * hf;
#pragma unroll
        for (int i = 0; i < 8; ++i) {
            const float4 v0 = src[2 * i], v1 = src[2 * i + 1];
            v8bf c;
            c[0] = (__bf16)v0.x; c[1] = (__bf16)v0.y; c[2] = (__bf16)v0.z; c[3] = (__bf16)v0.w;
            c[4] = (__bf16)v1.x; c[5] = (__bf16)v1.y; c[6] = (__bf16)v1.z; c[7] = (__bf16)v1.w;
            *(v8bf*)(dst + 8 * i) = c;
        }
    }
    __syncthreads();

    const int m0   = wave * 16;     // this wave's 16-edge stripe
    const int nIdx = lane & 15;
    const int hfm  = lane >> 4;

    // ---- Layer 1: x1 = relu(hA @ W1 + b1)   K=128, Nout=128 ----------------
    {
        v16bf a[4];
#pragma unroll
        for (int kt = 0; kt < 4; ++kt) a[kt] = load_frag_a(hA, HID, m0, 32 * kt, lane);
        for (int ni = 0; ni < 8; ++ni) {
            v8f acc = {};
#pragma unroll
            for (int kt = 0; kt < 4; ++kt) {
                v16bf b = load_frag_b(w1t, HID, ni * 16, 32 * kt, lane);
                acc = __builtin_amdgcn_wmma_f32_16x16x32_bf16(
                        false, a[kt], false, b, (short)0, acc, false, false);
            }
            const int n = ni * 16 + nIdx;
            const float bias = b1[n];
#pragma unroll
            for (int r = 0; r < 8; ++r) {
                float v = acc[r] + bias;
                v = v > 0.0f ? v : 0.0f;
                x1[(m0 + r + 8 * hfm) * HID + n] = (__bf16)v;
            }
        }
    }
    __syncthreads();

    // ---- Layer 2: x2 = relu(x1 @ W2 + b2)   K=128, Nout=256 ----------------
    {
        v16bf a[4];
#pragma unroll
        for (int kt = 0; kt < 4; ++kt) a[kt] = load_frag_a(x1, HID, m0, 32 * kt, lane);
        for (int ni = 0; ni < 16; ++ni) {
            v8f acc = {};
#pragma unroll
            for (int kt = 0; kt < 4; ++kt) {
                v16bf b = load_frag_b(w2t, HID, ni * 16, 32 * kt, lane);
                acc = __builtin_amdgcn_wmma_f32_16x16x32_bf16(
                        false, a[kt], false, b, (short)0, acc, false, false);
            }
            const int n = ni * 16 + nIdx;
            const float bias = b2[n];
#pragma unroll
            for (int r = 0; r < 8; ++r) {
                float v = acc[r] + bias;
                v = v > 0.0f ? v : 0.0f;
                x2[(m0 + r + 8 * hfm) * HID2 + n] = (__bf16)v;
            }
        }
    }
    __syncthreads();

    // ---- Layer 3: delta = x2 @ W3 + b3, scatter-add  K=256, Nout=128 -------
    {
        v16bf a[8];
#pragma unroll
        for (int kt = 0; kt < 8; ++kt) a[kt] = load_frag_a(x2, HID2, m0, 32 * kt, lane);
        for (int ni = 0; ni < 8; ++ni) {
            v8f acc = {};
#pragma unroll
            for (int kt = 0; kt < 8; ++kt) {
                v16bf b = load_frag_b(w3t, HID2, ni * 16, 32 * kt, lane);
                acc = __builtin_amdgcn_wmma_f32_16x16x32_bf16(
                        false, a[kt], false, b, (short)0, acc, false, false);
            }
            const int n = ni * 16 + nIdx;
            const float bias = b3[n];
#pragma unroll
            for (int r = 0; r < 8; ++r) {
                const int m = m0 + r + 8 * hfm;
                const int e = e0 + m;
                if (e < nEdges) {
                    const float v = acc[r] + bias;
                    float* dst = out + (size_t)srows[m] * NODE_DIM + n;
                    __hip_atomic_fetch_add(dst, v, __ATOMIC_RELAXED,
                                           __HIP_MEMORY_SCOPE_AGENT);
                }
            }
        }
    }
}

// ---------------------------------------------------------------------------
extern "C" void kernel_launch(void* const* d_in, const int* in_sizes, int n_in,
                              void* d_out, int out_size, void* d_ws, size_t ws_size,
                              hipStream_t stream) {
    const float* h  = (const float*)d_in[0];
    const int*   ei = (const int*)d_in[1];     // [2][E]; row = ei[0..E)
    const float* W1 = (const float*)d_in[2];
    const float* b1 = (const float*)d_in[3];
    const float* W2 = (const float*)d_in[4];
    const float* b2 = (const float*)d_in[5];
    const float* W3 = (const float*)d_in[6];
    const float* b3 = (const float*)d_in[7];
    float* out = (float*)d_out;

    const int E = in_sizes[1] / 2;

    // workspace: bf16 transposed weights (32 + 64 + 64 KB)
    __bf16* w1t = (__bf16*)d_ws;
    __bf16* w2t = w1t + 128 * 128;
    __bf16* w3t = w2t + 256 * 128;

    gnn_prep_weights<<<(256 * 128 + 255) / 256, 256, 0, stream>>>(W1, W2, W3,
                                                                  w1t, w2t, w3t);

    const int n4 = out_size / 4;
    gnn_init_out<<<(n4 + 255) / 256 < 4096 ? (n4 + 255) / 256 : 4096, 256, 0,
                 stream>>>((const float4*)h, (float4*)out, n4);

    const int nBlocks = (E + TILE_E - 1) / TILE_E;      // 12500 for E=1.6M
    const size_t shmem = (size_t)(TILE_E * HID + TILE_E * HID + TILE_E * HID2)
                             * sizeof(__bf16)
                         + TILE_E * sizeof(int);        // 131,584 B
    gnn_edge_mlp<<<nBlocks, 256, shmem, stream>>>(h, ei, w1t, w2t, w3t,
                                                  b1, b2, b3, out, E);
}
